// ContrastiveLoss_45320494907763
// MI455X (gfx1250) — matrix operand order repaired and checked by reading, not compiled, not run on hardware
//
#include <hip/hip_runtime.h>
#include <hip/hip_bf16.h>

// ---------------------------------------------------------------------------
// ContrastiveLoss for x[4096,1024] f32, y[4096] i64.
//   k1: row-normalize x -> xn (f16, workspace)                    [8 MB]
//   k2: S = xn * xn^T via v_wmma_f32_16x16x32_f16, staged through
//       LDS with double-buffered global_load_async_to_lds_b128    [64 MB, L2]
//   k3: per-row dissimilar sums + same-label loss terms
//   k4: deterministic final reduction -> d_out scalar
// ---------------------------------------------------------------------------

typedef __attribute__((ext_vector_type(16))) _Float16 v16h;
typedef __attribute__((ext_vector_type(8)))  float    v8f;
typedef __attribute__((ext_vector_type(4)))  _Float16 v4h;

namespace {
constexpr int   kN    = 4096;
constexpr int   kD    = 1024;
constexpr float kT    = 0.5f;
constexpr float kEps  = 1e-8f;
constexpr float kCap  = 1e10f;

constexpr int BM = 128, BN = 128, BK = 32;
constexpr int SA = 40;              // LDS row stride in halves (80 B: 16B-aligned, bank-skewed)

union Frag { v16h v; uint4 q[2]; }; // 32-byte WMMA operand fragment
}

// ---------------- Kernel 1: row normalize, f32 -> f16 ----------------------
__global__ __launch_bounds__(256)
void cl_normalize(const float* __restrict__ x, _Float16* __restrict__ xn) {
    const int row = blockIdx.x;
    const int tid = threadIdx.x;
    const float4 v = ((const float4*)(x + (size_t)row * kD))[tid]; // 256*4 = 1024
    float ss = v.x * v.x + v.y * v.y + v.z * v.z + v.w * v.w;
    #pragma unroll
    for (int off = 16; off > 0; off >>= 1) ss += __shfl_xor(ss, off);
    __shared__ float red[8];
    if ((tid & 31) == 0) red[tid >> 5] = ss;
    __syncthreads();
    float sum = 0.f;
    #pragma unroll
    for (int i = 0; i < 8; i++) sum += red[i];
    const float inv = 1.0f / fmaxf(sqrtf(sum), kEps);
    v4h h;
    h.x = (_Float16)(v.x * inv);
    h.y = (_Float16)(v.y * inv);
    h.z = (_Float16)(v.z * inv);
    h.w = (_Float16)(v.w * inv);
    *(v4h*)(xn + (size_t)row * kD + tid * 4) = h;
}

// ---------------- Kernel 2: S = xn * xn^T (WMMA f16 -> f32) ----------------
// Double-buffered LDS fed by async global->LDS DMA (ASYNCcnt), one barrier
// per K-step; WMMA compute overlaps the next slab's DMA.
__global__ __launch_bounds__(256)
void cl_gram_wmma(const _Float16* __restrict__ xn, float* __restrict__ S) {
    __shared__ alignas(16) _Float16 As[2][BM * SA];
    __shared__ alignas(16) _Float16 Bs[2][BN * SA];

    const int row0 = blockIdx.y * BM;
    const int col0 = blockIdx.x * BN;
    const int tid  = threadIdx.x;
    const int lane = tid & 31;
    const int w    = tid >> 5;          // 8 wave32 waves
    const int wm   = (w & 3) * 32;      // wave's M offset in tile (4 waves in M)
    const int wn   = (w >> 2) * 64;     // wave's N offset in tile (2 waves in N)

    // Staging: thread t owns 32 contiguous bytes (16 halves) of one tile row.
    const int lr = tid >> 1;            // 0..127: tile row
    const int lk = (tid & 1) * 16;      // 0 or 16: K half
    const _Float16* aptr = xn + (size_t)(row0 + lr) * kD + lk;
    const _Float16* bptr = xn + (size_t)(col0 + lr) * kD + lk;

    // LDS byte offsets (low 32 bits of the generic pointer == DS address).
    const unsigned ldsA[2] = { (unsigned)(size_t)&As[0][lr * SA + lk],
                               (unsigned)(size_t)&As[1][lr * SA + lk] };
    const unsigned ldsB[2] = { (unsigned)(size_t)&Bs[0][lr * SA + lk],
                               (unsigned)(size_t)&Bs[1][lr * SA + lk] };

    // Async DMA of one 32-byte chunk pair into LDS buffer `buf` for K-slab k0.
    auto issue = [&](int k0, int buf) {
        const unsigned long long ga = (unsigned long long)(size_t)(aptr + k0);
        const unsigned long long gb = (unsigned long long)(size_t)(bptr + k0);
        asm volatile("global_load_async_to_lds_b128 %0, %1, off"
                     :: "v"(ldsA[buf]), "v"(ga) : "memory");
        asm volatile("global_load_async_to_lds_b128 %0, %1, off offset:16"
                     :: "v"(ldsA[buf]), "v"(ga) : "memory");
        asm volatile("global_load_async_to_lds_b128 %0, %1, off"
                     :: "v"(ldsB[buf]), "v"(gb) : "memory");
        asm volatile("global_load_async_to_lds_b128 %0, %1, off offset:16"
                     :: "v"(ldsB[buf]), "v"(gb) : "memory");
    };

    // ISA fragment lane mapping (CDNA5 ISA 7.12.2):
    //   A 16x32 f16: lane L -> M=L&15; K chunks [8*(L>>4)..+7] and [16+8*(L>>4)..+7]
    //   B 32x16 f16: lane L -> N=L&15; K chunks [8*(L>>4)..+7] and [16+8*(L>>4)..+7]
    const int mrow = lane & 15;
    const int hi   = lane >> 4;
    const int ka   = hi * 8;
    const int kb   = hi * 8;

    v8f acc[2][4] = {};

    auto compute = [&](const _Float16* __restrict__ Ab,
                       const _Float16* __restrict__ Bb) {
        Frag a[2], b[4];
        #pragma unroll
        for (int i = 0; i < 2; i++) {
            const int r = wm + i * 16 + mrow;
            a[i].q[0] = *(const uint4*)&Ab[r * SA + ka];
            a[i].q[1] = *(const uint4*)&Ab[r * SA + 16 + ka];
        }
        #pragma unroll
        for (int j = 0; j < 4; j++) {
            const int c = wn + j * 16 + mrow;
            b[j].q[0] = *(const uint4*)&Bb[c * SA + kb];
            b[j].q[1] = *(const uint4*)&Bb[c * SA + kb + 16];
        }
        #pragma unroll
        for (int i = 0; i < 2; i++)
            #pragma unroll
            for (int j = 0; j < 4; j++)
                acc[i][j] = __builtin_amdgcn_wmma_f32_16x16x32_f16(
                    false, a[i].v, false, b[j].v, (short)0, acc[i][j], false, false);
    };

    issue(0, 0);                                    // prologue: fill buffer 0
    #pragma unroll 1
    for (int k0 = 0; k0 < kD; k0 += 2 * BK) {
        // ---- slab k0 in buffer 0 ----
        asm volatile("s_wait_asynccnt 0x0" ::: "memory"); // my DMA portions landed
        __syncthreads();                                  // everyone's landed
        issue(k0 + BK, 1);                                // overlap: fill buffer 1
        compute(As[0], Bs[0]);
        // ---- slab k0+BK in buffer 1 ----
        asm volatile("s_wait_asynccnt 0x0" ::: "memory");
        __syncthreads();
        if (k0 + 2 * BK < kD) issue(k0 + 2 * BK, 0);      // overlap: refill buffer 0
        compute(As[1], Bs[1]);
    }

    // Store raw Gram tile (C/D layout: lane -> N=lane&15, VGPR v -> M = v + 8*(lane>>4))
    const int colE   = lane & 15;
    const int rowSel = (lane >> 4) * 8;
    #pragma unroll
    for (int i = 0; i < 2; i++) {
        #pragma unroll
        for (int j = 0; j < 4; j++) {
            float* out = S + (size_t)(row0 + wm + i * 16 + rowSel) * kN
                           + (col0 + wn + j * 16 + colE);
            #pragma unroll
            for (int v = 0; v < 8; v++) out[(size_t)v * kN] = acc[i][j][v];
        }
    }
}

// ---------------- Kernel 3: per-row dissimilar-sum + loss ------------------
__global__ __launch_bounds__(256)
void cl_row_loss(const float* __restrict__ S, const long long* __restrict__ y,
                 float* __restrict__ rowLoss) {
    const int i   = blockIdx.x;
    const int tid = threadIdx.x;
    const long long yi = y[i];
    const float* row = S + (size_t)i * kN;

    // Pass 1: D_i = sum over j with y_j != y_i of capped exp-sim (diag auto-excluded)
    float dsum = 0.f;
    for (int j = tid; j < kN; j += 256) {
        float sim = fminf(fmaxf(row[j], -1.0f + kEps), 1.0f - kEps);
        float e   = fminf(expf(sim / kT), kCap);
        if (j == i) e = 0.f;
        if (y[j] != yi) dsum += e;
    }
    #pragma unroll
    for (int off = 16; off > 0; off >>= 1) dsum += __shfl_xor(dsum, off);
    __shared__ float red[8];
    if ((tid & 31) == 0) red[tid >> 5] = dsum;
    __syncthreads();
    float D = 0.f;
    #pragma unroll
    for (int k = 0; k < 8; k++) D += red[k];
    __syncthreads();

    // Pass 2: only same-label off-diagonal entries contribute:
    //   loss += -log(max(s/(s + D + eps), eps))
    float loss = 0.f;
    for (int j = tid; j < kN; j += 256) {
        if (j == i || y[j] != yi) continue;
        float sim = fminf(fmaxf(row[j], -1.0f + kEps), 1.0f - kEps);
        float e   = fminf(expf(sim / kT), kCap);
        float ratio = e / (e + D + kEps);
        loss += -logf(fmaxf(ratio, kEps));
    }
    #pragma unroll
    for (int off = 16; off > 0; off >>= 1) loss += __shfl_xor(loss, off);
    if ((tid & 31) == 0) red[tid >> 5] = loss;
    __syncthreads();
    if (tid == 0) {
        float t = 0.f;
        #pragma unroll
        for (int k = 0; k < 8; k++) t += red[k];
        rowLoss[i] = t;
    }
}

// ---------------- Kernel 4: deterministic final reduction ------------------
__global__ __launch_bounds__(256)
void cl_reduce(const float* __restrict__ rowLoss, float* __restrict__ out) {
    const int tid = threadIdx.x;
    float s = 0.f;
    for (int j = tid; j < kN; j += 256) s += rowLoss[j];
    #pragma unroll
    for (int off = 16; off > 0; off >>= 1) s += __shfl_xor(s, off);
    __shared__ float red[8];
    if ((tid & 31) == 0) red[tid >> 5] = s;
    __syncthreads();
    if (tid == 0) {
        float t = 0.f;
        #pragma unroll
        for (int k = 0; k < 8; k++) t += red[k];
        *out = t / (2.0f * (float)kN);
    }
}

extern "C" void kernel_launch(void* const* d_in, const int* in_sizes, int n_in,
                              void* d_out, int out_size, void* d_ws, size_t ws_size,
                              hipStream_t stream) {
    const float*     x = (const float*)d_in[0];
    const long long* y = (const long long*)d_in[1];
    float* out = (float*)d_out;

    // Workspace: xn f16 (8 MB) | S f32 (64 MB) | rowLoss f32 (16 KB)
    _Float16* xn = (_Float16*)d_ws;
    float* S       = (float*)((char*)d_ws + (size_t)kN * kD * sizeof(_Float16));
    float* rowLoss = (float*)((char*)S + (size_t)kN * kN * sizeof(float));

    cl_normalize<<<kN, 256, 0, stream>>>(x, xn);
    dim3 grid(kN / BN, kN / BM);
    cl_gram_wmma<<<grid, 256, 0, stream>>>(xn, S);
    cl_row_loss<<<kN, 256, 0, stream>>>(S, y, rowLoss);
    cl_reduce<<<1, 256, 0, stream>>>(rowLoss, out);
}
